// MultiHeadCausalAttention_8340826489273
// MI455X (gfx1250) — compile-verified, matrix operand order checked by reading, not run on hardware
//
#include <hip/hip_runtime.h>
#include <math.h>

#define BB 4
#define SS 2048
#define DD 1024
#define HH 16
#define HDD 64

typedef __attribute__((ext_vector_type(16))) __bf16 bf16x16;
typedef __attribute__((ext_vector_type(8)))  float  f32x8;
typedef __attribute__((ext_vector_type(8)))  unsigned short u16x8;
typedef __attribute__((ext_vector_type(16))) unsigned short u16x16;

// ---------- CDNA5 async global->LDS staging (ASYNCcnt path), guarded ----------
#if defined(__AMDGCN__) && __has_builtin(__builtin_amdgcn_global_load_async_to_lds_b128) && \
    __has_builtin(__builtin_amdgcn_s_wait_asynccnt)
#define USE_ASYNC_LDS 1
#else
#define USE_ASYNC_LDS 0
#endif

#if USE_ASYNC_LDS
// builtin signature (per hipcc diagnostic): pointee is GCC-vector "int vector_size(16)"
typedef int i32x4v __attribute__((vector_size(16)));
typedef __attribute__((address_space(1))) i32x4v* gptr_v4i;
typedef __attribute__((address_space(3))) i32x4v* lptr_v4i;

static __device__ __forceinline__ void async_cp16(const unsigned short* g,
                                                  unsigned short* l) {
  // 16 bytes per lane, memory -> LDS without touching VGPRs (ASYNCcnt tracked)
  __builtin_amdgcn_global_load_async_to_lds_b128((gptr_v4i)g, (lptr_v4i)l, 0, 0);
}
static __device__ __forceinline__ void async_wait0() {
  __builtin_amdgcn_s_wait_asynccnt(0);
}
#endif

// ---------- helpers ----------

static __device__ __forceinline__ unsigned short f2bf(float f) {
  unsigned int u = __builtin_bit_cast(unsigned int, f);
  unsigned int r = u + 0x7FFFu + ((u >> 16) & 1u);   // round-to-nearest-even
  if ((u & 0x7F800000u) == 0x7F800000u) r = u;        // NaN/Inf passthrough
  return (unsigned short)(r >> 16);
}

static __device__ __forceinline__ u16x8 ld8(const unsigned short* p) {
  return *(const u16x8*)p;
}

static __device__ __forceinline__ bf16x16 frag16(u16x8 lo, u16x8 hi) {
  u16x16 t;
#pragma unroll
  for (int i = 0; i < 8; ++i) { t[i] = lo[i]; t[i + 8] = hi[i]; }
  return __builtin_bit_cast(bf16x16, t);
}

static __device__ __forceinline__ f32x8 wmma_bf16(bf16x16 a, bf16x16 b, f32x8 c) {
  return __builtin_amdgcn_wmma_f32_16x16x32_bf16(false, a, false, b, (short)0, c,
                                                 false, false);
}

// ---------- conversion kernels ----------

__global__ void cvt_bf16_kernel(const float* __restrict__ src,
                                unsigned short* __restrict__ dst, int n) {
  int i = blockIdx.x * blockDim.x + threadIdx.x;
  if (i < n) dst[i] = f2bf(src[i]);
}

// src: [K][N] row-major f32 -> dst: [N][K] bf16 (transposed)
__global__ void cvt_bf16_T_kernel(const float* __restrict__ src,
                                  unsigned short* __restrict__ dst, int K, int N) {
  int i = blockIdx.x * blockDim.x + threadIdx.x;
  if (i < K * N) {
    int k = i / N, n = i % N;
    dst[(size_t)n * K + k] = f2bf(src[i]);
  }
}

// ---------- shared GEMM core: C[128x128] += A[128xKT] * BT[128xKT]^T ----------
// A row-major bf16 [M][KT]; BT row-major bf16 [N][KT]. 256 threads = 8 waves.
// Wave (wm,wn) in 2x4 grid owns 64x32; acc[mi][ni] are 16x16 f32 tiles.

#define LSTR 40   // LDS row stride in elements (padded; 80 bytes, 16B-aligned)

template <int KT>
static __device__ __forceinline__ void gemm_core(const unsigned short* __restrict__ A,
                                                 const unsigned short* __restrict__ BT,
                                                 unsigned short* Alds,
                                                 unsigned short* Blds,
                                                 int m0, int n0, f32x8 acc[4][2]) {
  const int tid  = threadIdx.x;
  const int lane = tid & 31;
  const int wv   = tid >> 5;
  const int wm   = wv & 1;
  const int wn   = wv >> 1;
  const int half = lane >> 4;
  const int ln   = lane & 15;
  const int kb8  = half * 8;      // per-lane K base within fragment
  const int ldr  = tid >> 1;      // staging row 0..127
  const int ldh  = (tid & 1) * 16;

  const f32x8 zero = {0.f, 0.f, 0.f, 0.f, 0.f, 0.f, 0.f, 0.f};
#pragma unroll
  for (int mi = 0; mi < 4; ++mi)
#pragma unroll
    for (int ni = 0; ni < 2; ++ni) acc[mi][ni] = zero;

  for (int k0 = 0; k0 < KT; k0 += 32) {
    __syncthreads();
    const unsigned short* ag = A + (size_t)(m0 + ldr) * KT + k0 + ldh;
    const unsigned short* bg = BT + (size_t)(n0 + ldr) * KT + k0 + ldh;
#if USE_ASYNC_LDS
    async_cp16(ag,     &Alds[ldr * LSTR + ldh]);
    async_cp16(ag + 8, &Alds[ldr * LSTR + ldh + 8]);
    async_cp16(bg,     &Blds[ldr * LSTR + ldh]);
    async_cp16(bg + 8, &Blds[ldr * LSTR + ldh + 8]);
    async_wait0();
#else
    *(u16x8*)&Alds[ldr * LSTR + ldh]     = ld8(ag);
    *(u16x8*)&Alds[ldr * LSTR + ldh + 8] = ld8(ag + 8);
    *(u16x8*)&Blds[ldr * LSTR + ldh]     = ld8(bg);
    *(u16x8*)&Blds[ldr * LSTR + ldh + 8] = ld8(bg + 8);
#endif
    __syncthreads();

    bf16x16 af[4], bf[2];
#pragma unroll
    for (int mi = 0; mi < 4; ++mi) {
      int row = wm * 64 + mi * 16 + ln;
      af[mi] = frag16(ld8(&Alds[row * LSTR + kb8]),
                      ld8(&Alds[row * LSTR + kb8 + 16]));
    }
#pragma unroll
    for (int ni = 0; ni < 2; ++ni) {
      int col = wn * 32 + ni * 16 + ln;
      bf[ni] = frag16(ld8(&Blds[col * LSTR + kb8]),
                      ld8(&Blds[col * LSTR + kb8 + 16]));
    }
#pragma unroll
    for (int mi = 0; mi < 4; ++mi)
#pragma unroll
      for (int ni = 0; ni < 2; ++ni)
        acc[mi][ni] = wmma_bf16(af[mi], bf[ni], acc[mi][ni]);
  }
}

// ---------- kernel 1: QKV projection + bias + scatter to [B,H,S,64] bf16 ----------

__global__ __launch_bounds__(256) void gemm_qkv_kernel(
    const unsigned short* __restrict__ xb, const unsigned short* __restrict__ wqT,
    const float* __restrict__ bias, unsigned short* __restrict__ qb,
    unsigned short* __restrict__ kb, unsigned short* __restrict__ vb) {
  __shared__ unsigned short Alds[128 * LSTR];
  __shared__ unsigned short Blds[128 * LSTR];
  const int n0 = blockIdx.x * 128;
  const int m0 = blockIdx.y * 128;

  f32x8 acc[4][2];
  gemm_core<DD>(xb, wqT, Alds, Blds, m0, n0, acc);

  const int lane = threadIdx.x & 31;
  const int wv   = threadIdx.x >> 5;
  const int wm = wv & 1, wn = wv >> 1;
  const int half = lane >> 4, ln = lane & 15;

#pragma unroll
  for (int mi = 0; mi < 4; ++mi) {
#pragma unroll
    for (int ni = 0; ni < 2; ++ni) {
      const int n = n0 + wn * 32 + ni * 16 + ln;
      const float bs   = bias[n];
      const int which  = n >> 10;          // 0=Q 1=K 2=V
      const int h      = (n >> 6) & 15;
      const int hd     = n & 63;
      unsigned short* dst = (which == 0) ? qb : ((which == 1) ? kb : vb);
      const float sc = (which == 0) ? 0.125f : 1.0f;   // 1/sqrt(HD) folded into Q
#pragma unroll
      for (int r = 0; r < 8; ++r) {
        const int m = m0 + wm * 64 + mi * 16 + r + half * 8;
        const int b = m >> 11;             // m = b*S + s, S=2048
        const int s = m & 2047;
        dst[(((size_t)b * HH + h) * SS + s) * HDD + hd] = f2bf((acc[mi][ni][r] + bs) * sc);
      }
    }
  }
}

// ---------- kernel 2: flash-style causal attention ----------
// block = 256 threads = 8 waves; block handles 128 queries of one (b,h).

__global__ __launch_bounds__(256) void attn_kernel(
    const unsigned short* __restrict__ qbuf, const unsigned short* __restrict__ kbuf,
    const unsigned short* __restrict__ vbuf, unsigned short* __restrict__ att) {
  __shared__ unsigned short Klds[32 * 72];       // [key][hd], padded
  __shared__ unsigned short VT[64 * LSTR];       // [hd][key], padded
  __shared__ unsigned short Plds[128 * LSTR];    // per-wave 16x32 P tiles

  const int QBLK = SS / 128;                     // 16
  const int bh   = blockIdx.x / QBLK;
  const int qblk = blockIdx.x % QBLK;
  const int b = bh / HH, h = bh % HH;
  const int q0b = qblk * 128;

  const int tid = threadIdx.x, lane = tid & 31, wv = tid >> 5;
  const int half = lane >> 4, ln = lane & 15;
  const int kb8 = half * 8;
  const int q0 = q0b + wv * 16;

  // Q fragments (A-layout, K = head dim), loaded once, pre-scaled by 1/8 already
  const unsigned short* Qrow = qbuf + ((size_t)bh * SS + q0 + ln) * HDD;
  bf16x16 qf[2];
#pragma unroll
  for (int c = 0; c < 2; ++c)
    qf[c] = frag16(ld8(Qrow + c * 32 + kb8), ld8(Qrow + c * 32 + kb8 + 16));

  const f32x8 zero = {0.f, 0.f, 0.f, 0.f, 0.f, 0.f, 0.f, 0.f};
  f32x8 O[4];
#pragma unroll
  for (int nt = 0; nt < 4; ++nt) O[nt] = zero;
  float mrow[8], lrow[8];
#pragma unroll
  for (int r = 0; r < 8; ++r) { mrow[r] = -1e30f; lrow[r] = 0.f; }

  const unsigned short* Kb = kbuf + (size_t)bh * SS * HDD;
  const unsigned short* Vb = vbuf + (size_t)bh * SS * HDD;
  const int nkt = q0b / 32 + 4;                  // uniform per block (barrier-safe)

  const int ldr = tid >> 3;                      // 0..31  staging key row
  const int ldc = (tid & 7) * 8;                 // 0..56  staging hd col group

  for (int it = 0; it < nkt; ++it) {
    const int kb0 = it * 32;
    __syncthreads();
    // stage K tile [32][64] row-major (padded to 72): async global->LDS path
#if USE_ASYNC_LDS
    async_cp16(Kb + (size_t)(kb0 + ldr) * HDD + ldc, &Klds[ldr * 72 + ldc]);
#else
    *(u16x8*)&Klds[ldr * 72 + ldc] = ld8(Kb + (size_t)(kb0 + ldr) * HDD + ldc);
#endif
    // stage V tile transposed -> VT[hd][key] (needs element scatter, VGPR path)
    u16x8 vvv = ld8(Vb + (size_t)(kb0 + ldr) * HDD + ldc);
#pragma unroll
    for (int i = 0; i < 8; ++i) VT[(ldc + i) * LSTR + ldr] = vvv[i];
#if USE_ASYNC_LDS
    async_wait0();
#endif
    __syncthreads();

    // scores: S^tile = Q(16x64) x K^T ; two 16-key subtiles, 2 chained WMMAs each
    f32x8 st[2];
#pragma unroll
    for (int t = 0; t < 2; ++t) {
      f32x8 s = zero;
#pragma unroll
      for (int c = 0; c < 2; ++c) {
        const int key = t * 16 + ln;
        bf16x16 kf = frag16(ld8(&Klds[key * 72 + c * 32 + kb8]),
                            ld8(&Klds[key * 72 + c * 32 + kb8 + 16]));
        s = wmma_bf16(qf[c], kf, s);
      }
      st[t] = s;
    }

    // online softmax per query row (rows live in one 16-lane half)
#pragma unroll
    for (int r = 0; r < 8; ++r) {
      const int qg  = q0 + r + half * 8;
      const int kg0 = kb0 + ln;
      const int kg1 = kb0 + 16 + ln;
      float s0 = st[0][r]; if (kg0 > qg) s0 = -1e30f;
      float s1 = st[1][r]; if (kg1 > qg) s1 = -1e30f;
      float rm = fmaxf(s0, s1);
      rm = fmaxf(rm, __shfl_xor(rm, 1, 32));
      rm = fmaxf(rm, __shfl_xor(rm, 2, 32));
      rm = fmaxf(rm, __shfl_xor(rm, 4, 32));
      rm = fmaxf(rm, __shfl_xor(rm, 8, 32));
      const float mn   = fmaxf(mrow[r], rm);
      const float corr = __expf(mrow[r] - mn);
      const float p0 = __expf(s0 - mn);
      const float p1 = __expf(s1 - mn);
      float ps = p0 + p1;
      ps += __shfl_xor(ps, 1, 32);
      ps += __shfl_xor(ps, 2, 32);
      ps += __shfl_xor(ps, 4, 32);
      ps += __shfl_xor(ps, 8, 32);
      lrow[r] = lrow[r] * corr + ps;
      mrow[r] = mn;
#pragma unroll
      for (int nt = 0; nt < 4; ++nt) O[nt][r] *= corr;
      // write P (bf16) into this wave's private LDS tile [16][32]
      const int prow = (wv * 16 + r + half * 8) * LSTR;
      Plds[prow + ln]      = f2bf(p0);
      Plds[prow + 16 + ln] = f2bf(p1);
    }

    // PV: A = P(16x32) from LDS; B = V(32keys x 64hd) via VT LDS; 4 WMMAs
    bf16x16 pf = frag16(ld8(&Plds[(wv * 16 + ln) * LSTR + kb8]),
                        ld8(&Plds[(wv * 16 + ln) * LSTR + kb8 + 16]));
#pragma unroll
    for (int nt = 0; nt < 4; ++nt) {
      const int hd = nt * 16 + ln;
      bf16x16 vf = frag16(ld8(&VT[hd * LSTR + kb8]),
                          ld8(&VT[hd * LSTR + kb8 + 16]));
      O[nt] = wmma_bf16(pf, vf, O[nt]);
    }
  }

  // epilogue: O /= l, store bf16 to att[B,S, h*64+hd] (= [B*S, D] row-major)
#pragma unroll
  for (int r = 0; r < 8; ++r) {
    const float rl = 1.0f / lrow[r];
    const int qg = q0 + r + half * 8;
#pragma unroll
    for (int nt = 0; nt < 4; ++nt) {
      const int col = h * HDD + nt * 16 + ln;
      att[((size_t)b * SS + qg) * DD + col] = f2bf(O[nt][r] * rl);
    }
  }
}

// ---------- kernel 3: output projection + bias -> f32 ----------

__global__ __launch_bounds__(256) void gemm_out_kernel(
    const unsigned short* __restrict__ attb, const unsigned short* __restrict__ woT,
    const float* __restrict__ bias, float* __restrict__ out) {
  __shared__ unsigned short Alds[128 * LSTR];
  __shared__ unsigned short Blds[128 * LSTR];
  const int n0 = blockIdx.x * 128;
  const int m0 = blockIdx.y * 128;

  f32x8 acc[4][2];
  gemm_core<DD>(attb, woT, Alds, Blds, m0, n0, acc);

  const int lane = threadIdx.x & 31;
  const int wv   = threadIdx.x >> 5;
  const int wm = wv & 1, wn = wv >> 1;
  const int half = lane >> 4, ln = lane & 15;

#pragma unroll
  for (int mi = 0; mi < 4; ++mi) {
#pragma unroll
    for (int ni = 0; ni < 2; ++ni) {
      const int n = n0 + wn * 32 + ni * 16 + ln;
      const float bs = bias[n];
#pragma unroll
      for (int r = 0; r < 8; ++r) {
        const int m = m0 + wm * 64 + mi * 16 + r + half * 8;
        out[(size_t)m * DD + n] = acc[mi][ni][r] + bs;
      }
    }
  }
}

// ---------- launch ----------

extern "C" void kernel_launch(void* const* d_in, const int* in_sizes, int n_in,
                              void* d_out, int out_size, void* d_ws, size_t ws_size,
                              hipStream_t stream) {
  (void)in_sizes; (void)n_in; (void)out_size; (void)ws_size;
  const float* x    = (const float*)d_in[0];
  const float* Wqkv = (const float*)d_in[1];
  const float* bqkv = (const float*)d_in[2];
  const float* Wout = (const float*)d_in[3];
  const float* bout = (const float*)d_in[4];
  float* out = (float*)d_out;

  char* ws = (char*)d_ws;
  const size_t XB = (size_t)BB * SS * DD * 2;            // 16 MB each
  unsigned short* xb   = (unsigned short*)ws;  ws += XB;
  unsigned short* wqT  = (unsigned short*)ws;  ws += (size_t)3 * DD * DD * 2;
  unsigned short* woT  = (unsigned short*)ws;  ws += (size_t)DD * DD * 2;
  unsigned short* qb   = (unsigned short*)ws;  ws += XB;
  unsigned short* kb   = (unsigned short*)ws;  ws += XB;
  unsigned short* vb   = (unsigned short*)ws;  ws += XB;
  unsigned short* attb = (unsigned short*)ws;  ws += XB;

  const int nX = BB * SS * DD;
  cvt_bf16_kernel<<<(nX + 255) / 256, 256, 0, stream>>>(x, xb, nX);
  const int nWq = DD * 3 * DD;
  cvt_bf16_T_kernel<<<(nWq + 255) / 256, 256, 0, stream>>>(Wqkv, wqT, DD, 3 * DD);
  const int nWo = DD * DD;
  cvt_bf16_T_kernel<<<(nWo + 255) / 256, 256, 0, stream>>>(Wout, woT, DD, DD);

  dim3 g1(3 * DD / 128, BB * SS / 128);        // 24 x 64
  gemm_qkv_kernel<<<g1, 256, 0, stream>>>(xb, wqT, bqkv, qb, kb, vb);

  attn_kernel<<<BB * HH * (SS / 128), 256, 0, stream>>>(qb, kb, vb, attb);

  dim3 g2(DD / 128, BB * SS / 128);            // 8 x 64
  gemm_out_kernel<<<g2, 256, 0, stream>>>(attb, woT, bout, out);
}
